// MultiHeadAttentionLayer_69982197121735
// MI455X (gfx1250) — matrix operand order repaired and checked by reading, not compiled, not run on hardware
//
#include <hip/hip_runtime.h>

#define NN   20000
#define E1   320000
#define E2N  1280000
#define HD   128
#define NHEAD 8

// NOTE: NN and E1 are both multiples of 16, so GEMM row tiles are always full.

typedef __attribute__((ext_vector_type(16))) _Float16 v16h;
typedef __attribute__((ext_vector_type(8)))  _Float16 v8h;
typedef __attribute__((ext_vector_type(8)))  float    v8f;

// ---- convert W[k][n] (fp32) -> Wt[w][n][k] (f16, transposed) ----
__global__ void convert_weights_k(const float* __restrict__ Wq, const float* __restrict__ Wk,
                                  const float* __restrict__ Wv, const float* __restrict__ We,
                                  _Float16* __restrict__ Wt) {
  int gid = blockIdx.x * blockDim.x + threadIdx.x;
  if (gid >= 4 * 128 * 128) return;
  int w = gid >> 14, r = gid & 16383;
  int n = r >> 7, k = r & 127;
  const float* src = (w == 0) ? Wq : (w == 1) ? Wk : (w == 2) ? Wv : We;
  Wt[gid] = (_Float16)src[k * 128 + n];
  (void)n;
}

__global__ void zero_k(float* __restrict__ p, long long cnt) {
  long long gid = (long long)blockIdx.x * blockDim.x + threadIdx.x;
  if (gid < cnt) p[gid] = 0.0f;
}

// ---- Out[M,128] = X[M,128] @ W + bias, WMMA f32_16x16x32_f16 ----
// block: 256 threads = 8 waves; wave w computes n-tile w; block row-tile of 16.
// M must be a multiple of 16 (true for NN and E1) -> no bounds checks.
__global__ void gemm128_wmma_k(const float* __restrict__ X, const _Float16* __restrict__ Wt,
                               const float* __restrict__ bias, float* __restrict__ Out) {
  __shared__ __align__(16) _Float16 ldsA[16 * 136];
  const int tid = threadIdx.x;
  const int rowBase = blockIdx.x * 16;
  {
    int r = tid >> 4;
    int c = (tid & 15) << 3;
    const float* xp = X + (size_t)(rowBase + r) * 128 + c;
    _Float16* lp = ldsA + r * 136 + c;
#pragma unroll
    for (int i = 0; i < 8; i++) lp[i] = (_Float16)xp[i];
  }
  __syncthreads();

  const int lane = tid & 31;
  const int wave = tid >> 5;      // n-tile 0..7
  const int m    = lane & 15;     // A row / B column within tile
  const int hs   = lane >> 4;     // half-wave select

  v8f acc = {};
#pragma unroll
  for (int t = 0; t < 4; t++) {
    // A fragment: 16 contiguous k per half-wave (lanes0-15 K 0..15, lanes16-31 K 16..31), matching B
    int k0 = t * 32 + hs * 16;
    v8h alo = *(const v8h*)(ldsA + m * 136 + k0);
    v8h ahi = *(const v8h*)(ldsA + m * 136 + k0 + 8);
    v16h a;
#pragma unroll
    for (int i = 0; i < 8; i++) { a[i] = alo[i]; a[i + 8] = ahi[i]; }
    // B fragment: lane = column n, 16 contiguous k per half-wave (Wt is [n][k])
    v16h b = *(const v16h*)(Wt + (size_t)(wave * 16 + m) * 128 + t * 32 + hs * 16);
    acc = __builtin_amdgcn_wmma_f32_16x16x32_f16(false, a, false, b, (short)0, acc, false, false);
  }

  const int col = wave * 16 + m;
  const float bv = bias[col];
  float* op = Out + (size_t)(rowBase + hs * 8) * 128 + col;
#pragma unroll
  for (int r = 0; r < 8; r++) {
    // C/D layout: VGPR r -> m = r + 8*(lane/16), n = lane%16
    op[(size_t)r * 128] = acc[r] + bv;
  }
}

// ---- sparse edge-conditioned attention; eout holds proj_e on entry, score on exit ----
__global__ void edge_sparse_k(const float* __restrict__ Q, const float* __restrict__ K,
                              const float* __restrict__ V,
                              const int* __restrict__ src, const int* __restrict__ dst,
                              float* __restrict__ eout,
                              float* __restrict__ wV, float* __restrict__ z) {
  long long gid = (long long)blockIdx.x * blockDim.x + threadIdx.x;
  if (gid >= (long long)E1 * HD) return;
  int edge = (int)(gid >> 7);
  int hd   = (int)(gid & 127);
  int s = src[edge], d = dst[edge];
  float score = K[(size_t)s * HD + hd] * Q[(size_t)d * HD + hd] * 0.25f * eout[gid];
  eout[gid] = score;
  float sum = score;
  sum += __shfl_xor(sum, 1);
  sum += __shfl_xor(sum, 2);
  sum += __shfl_xor(sum, 4);
  sum += __shfl_xor(sum, 8);
  float sE = expf(fminf(fmaxf(sum, -5.0f), 5.0f));
  atomicAdd(&wV[(size_t)d * HD + hd], V[(size_t)s * HD + hd] * sE);
  if ((hd & 15) == 0) atomicAdd(&z[d * NHEAD + (hd >> 4)], sE);
}

// ---- full graph: gated dot-exp with 3d relative position ----
__global__ void edge_full_k(const float* __restrict__ Q, const float* __restrict__ K,
                            const float* __restrict__ V,
                            const int* __restrict__ src2, const int* __restrict__ dst2,
                            const float* __restrict__ adj2, const float* __restrict__ rel,
                            float* __restrict__ wV2, float* __restrict__ z2) {
  long long gid = (long long)blockIdx.x * blockDim.x + threadIdx.x;
  if (gid >= (long long)E2N * HD) return;
  int edge = (int)(gid >> 7);
  int hd   = (int)(gid & 127);
  int s = src2[edge], d = dst2[edge];
  float prod = K[(size_t)s * HD + hd] * Q[(size_t)d * HD + hd] * 0.25f;
  float sum = prod;
  sum += __shfl_xor(sum, 1);
  sum += __shfl_xor(sum, 2);
  sum += __shfl_xor(sum, 4);
  sum += __shfl_xor(sum, 8);
  float val = expf(fminf(fmaxf(sum + rel[edge], -5.0f), 5.0f) * adj2[edge]);
  atomicAdd(&wV2[(size_t)d * HD + hd], V[(size_t)s * HD + hd] * val);
  if ((hd & 15) == 0) atomicAdd(&z2[d * NHEAD + (hd >> 4)], val);
}

__global__ void finalize_k(const float* __restrict__ wV, const float* __restrict__ z,
                           const float* __restrict__ wV2, const float* __restrict__ z2,
                           float* __restrict__ hout) {
  int gid = blockIdx.x * blockDim.x + threadIdx.x;
  if (gid >= NN * HD) return;
  int n = gid >> 7, h = (gid >> 4) & 7;
  hout[gid] = wV[gid]  / (z[n * NHEAD + h]  + 1e-6f)
            + wV2[gid] / (z2[n * NHEAD + h] + 1e-6f);
}

extern "C" void kernel_launch(void* const* d_in, const int* in_sizes, int n_in,
                              void* d_out, int out_size, void* d_ws, size_t ws_size,
                              hipStream_t stream) {
  const float* h    = (const float*)d_in[0];
  const float* e    = (const float*)d_in[1];
  const int*   src  = (const int*)d_in[2];
  const int*   dst  = (const int*)d_in[3];
  const int*   src2 = (const int*)d_in[4];
  const int*   dst2 = (const int*)d_in[5];
  const float* adj2 = (const float*)d_in[6];
  const float* rel  = (const float*)d_in[7];
  const float* Wq = (const float*)d_in[8];   const float* bq = (const float*)d_in[9];
  const float* Wk = (const float*)d_in[10];  const float* bk = (const float*)d_in[11];
  const float* Wv = (const float*)d_in[12];  const float* bv = (const float*)d_in[13];
  const float* We = (const float*)d_in[14];  const float* be = (const float*)d_in[15];

  float* out  = (float*)d_out;
  float* hout = out;
  float* eout = out + (size_t)NN * HD;       // proj_e staged here, overwritten with e_out

  float* Qm  = (float*)d_ws;                 // N*128
  float* Km  = Qm  + (size_t)NN * HD;
  float* Vm  = Km  + (size_t)NN * HD;
  float* wV  = Vm  + (size_t)NN * HD;        // zeroed region start
  float* wV2 = wV  + (size_t)NN * HD;
  float* z   = wV2 + (size_t)NN * HD;        // N*8
  float* z2  = z   + (size_t)NN * NHEAD;
  _Float16* Wt = (_Float16*)(z2 + (size_t)NN * NHEAD);  // 4*128*128 halfs, 32B aligned

  convert_weights_k<<<(4 * 128 * 128 + 255) / 256, 256, 0, stream>>>(Wq, Wk, Wv, We, Wt);

  long long zcnt = (long long)NN * (2 * HD + 2 * NHEAD);
  zero_k<<<(int)((zcnt + 255) / 256), 256, 0, stream>>>(wV, zcnt);

  gemm128_wmma_k<<<NN / 16, 256, 0, stream>>>(h, Wt,             bq, Qm);
  gemm128_wmma_k<<<NN / 16, 256, 0, stream>>>(h, Wt + 16384,     bk, Km);
  gemm128_wmma_k<<<NN / 16, 256, 0, stream>>>(h, Wt + 2 * 16384, bv, Vm);
  gemm128_wmma_k<<<E1 / 16, 256, 0, stream>>>(e, Wt + 3 * 16384, be, eout);

  long long t1 = (long long)E1 * HD;
  edge_sparse_k<<<(int)((t1 + 255) / 256), 256, 0, stream>>>(Qm, Km, Vm, src, dst, eout, wV, z);

  long long t2 = (long long)E2N * HD;
  edge_full_k<<<(int)((t2 + 255) / 256), 256, 0, stream>>>(Qm, Km, Vm, src2, dst2, adj2, rel, wV2, z2);

  finalize_k<<<(NN * HD + 255) / 256, 256, 0, stream>>>(wV, z, wV2, z2, hout);
}
